// Head_23751169146962
// MI455X (gfx1250) — matrix operand order repaired and checked by reading, not compiled, not run on hardware
//
#include <hip/hip_runtime.h>
#include <hip/hip_bf16.h>
#include <math.h>

// ---------------------------------------------------------------------------
// Single-head causal attention with fused QKV projection for MI455X (gfx1250).
//   x:[8,2048,1024]f32, Wq/Wk/Wv:[1024,64]f32 -> out:[8,2048,64]f32
// Kernel 1: q/k/v = x@W (bf16 WMMA 16x16x32, f32 accum), q pre-scaled 1/8,
//           stored bf16 in workspace (6 MB, L2-resident).
// Kernel 2: flash attention; K/V tiles staged with GLOBAL_LOAD_ASYNC_TO_LDS
//           (ASYNCcnt), V load overlapped with score GEMM + softmax.
// ---------------------------------------------------------------------------

typedef __attribute__((ext_vector_type(16))) __bf16 v16bf;
typedef __attribute__((ext_vector_type(8)))  float  v8f;

#define EMBED 1024
#define HDIM  64
#define SEQ   2048
#define BATCH 8
#define ROWS  (BATCH * SEQ)

union Frag {
    v16bf v;
    unsigned int w[8];
};

__device__ __forceinline__ unsigned short f2bf(float f) {
    __bf16 h = (__bf16)f;
    return __builtin_bit_cast(unsigned short, h);
}

__device__ __forceinline__ v8f vzero8() {
    v8f z = {0.f, 0.f, 0.f, 0.f, 0.f, 0.f, 0.f, 0.f};
    return z;
}

// DS ops are in-order within a wave; the wait + memory clobber also stops the
// compiler from reordering stage-stores vs gathers (and async issue).
__device__ __forceinline__ void lds_fence() {
    asm volatile("s_wait_dscnt 0" ::: "memory");
}

// A-fragment (16x32 bf16) K index for element pair j (contiguous layout):
// VGPR j holds K = (lane/16)*16 + j*2 + {0,1}
__device__ __forceinline__ int a_kidx(int j, int half) {
    return (half << 4) + (j << 1);
}

// Async global->LDS copy of 16 bytes (tracked by ASYNCcnt).
__device__ __forceinline__ void async_b128(unsigned lds_addr,
                                           unsigned long long gaddr) {
    asm volatile("global_load_async_to_lds_b128 %0, %1, off"
                 :: "v"(lds_addr), "v"(gaddr) : "memory");
}

// ---------------------------------------------------------------------------
// Kernel 1: QKV projection. 128 blocks x 256 threads; wave w handles rows
// [block*128 + w*16, +16). x read once; all three projections share the A tile.
// ---------------------------------------------------------------------------
__global__ __launch_bounds__(256) void qkv_proj_kernel(
    const float* __restrict__ x,
    const float* __restrict__ Wq,
    const float* __restrict__ Wk,
    const float* __restrict__ Wv,
    unsigned short* __restrict__ qb,
    unsigned short* __restrict__ kb,
    unsigned short* __restrict__ vb)
{
    __shared__ unsigned short sW[3 * 64 * 34];   // [mat][d(N)][k] stride 34 (padded)
    __shared__ unsigned short sX[8 * 16 * 34];   // [wave][m][k] stride 34 (padded)

    const int tid  = threadIdx.x;
    const int lane = tid & 31;
    const int wv   = tid >> 5;
    const int n    = lane & 15;
    const int half = lane >> 4;
    const int row0 = blockIdx.x * 128 + wv * 16;

    const float* Wm[3] = {Wq, Wk, Wv};

    v8f acc[3][4];
#pragma unroll
    for (int m3 = 0; m3 < 3; ++m3)
#pragma unroll
        for (int nt = 0; nt < 4; ++nt) acc[m3][nt] = vzero8();

    for (int kt = 0; kt < EMBED / 32; ++kt) {
        // --- stage weight k-tile (3 x 32 x 64 f32 -> bf16, transposed [d][k])
#pragma unroll
        for (int t = 0; t < 24; ++t) {
            int i   = tid + t * 256;        // 0 .. 6143
            int m3  = i >> 11;              // matrix
            int rem = i & 2047;
            int e   = rem >> 6;             // k within tile (0..31)
            int d   = rem & 63;             // output dim
            float val = Wm[m3][(size_t)(kt * 32 + e) * HDIM + d];
            sW[m3 * (64 * 34) + d * 34 + e] = f2bf(val);
        }
        // --- stage this wave's x tile (16 rows x 32 k), coalesced 128B rows
#pragma unroll
        for (int j = 0; j < 16; ++j) {
            float val = x[(size_t)(row0 + j) * EMBED + kt * 32 + lane];
            sX[wv * (16 * 34) + j * 34 + lane] = f2bf(val);
        }
        __syncthreads();

        // --- build A fragment for this wave's 16 rows
        Frag a;
        {
            const unsigned short* src = &sX[wv * (16 * 34) + n * 34];
#pragma unroll
            for (int j = 0; j < 8; ++j)
                a.w[j] = *(const unsigned int*)&src[a_kidx(j, half)];
        }
        // --- 12 WMMAs: 3 matrices x 4 N-tiles, K accumulated over tiles
#pragma unroll
        for (int m3 = 0; m3 < 3; ++m3) {
#pragma unroll
            for (int nt = 0; nt < 4; ++nt) {
                Frag b;
                const unsigned short* src =
                    &sW[m3 * (64 * 34) + (nt * 16 + n) * 34 + half * 16];
#pragma unroll
                for (int j = 0; j < 8; ++j)
                    b.w[j] = *(const unsigned int*)&src[2 * j];
                acc[m3][nt] = __builtin_amdgcn_wmma_f32_16x16x32_bf16(
                    false, a.v, false, b.v, (short)0, acc[m3][nt], false, false);
            }
        }
        __syncthreads();
    }

    // --- write q (scaled by 1/sqrt(64)=0.125), k, v as bf16 [row][64]
    unsigned short* dsts[3] = {qb, kb, vb};
#pragma unroll
    for (int m3 = 0; m3 < 3; ++m3) {
        const float sc = (m3 == 0) ? 0.125f : 1.0f;
#pragma unroll
        for (int nt = 0; nt < 4; ++nt)
#pragma unroll
            for (int r = 0; r < 8; ++r) {
                int row = row0 + r + 8 * half;
                dsts[m3][(size_t)row * HDIM + nt * 16 + n] =
                    f2bf(acc[m3][nt][r] * sc);
            }
    }
}

// ---------------------------------------------------------------------------
// Kernel 2: causal flash attention. 256 blocks x 128 threads (4 waves); wave
// w owns query tile q0 = (block%32)*64 + w*16 of batch block/32. No block
// barriers (divergent causal trip counts) -- staging is per-wave: K/V tiles
// arrive via GLOBAL_LOAD_ASYNC_TO_LDS_B128; V overlaps the score GEMM.
// ---------------------------------------------------------------------------
__global__ __launch_bounds__(128) void attn_kernel(
    const unsigned short* __restrict__ qb,
    const unsigned short* __restrict__ kb,
    const unsigned short* __restrict__ vb,
    float* __restrict__ out)
{
    // Unpadded [32 keys][64 d] tiles: 128B rows, 16B-aligned for async B128.
    __shared__ __align__(16) unsigned short sK[4 * 32 * 64];
    __shared__ __align__(16) unsigned short sV[4 * 32 * 64];
    __shared__ unsigned short sP[4 * 16 * 34];   // P / Q staging, stride 34

    const int tid   = threadIdx.x;
    const int lane  = tid & 31;
    const int wv    = tid >> 5;
    const int n     = lane & 15;
    const int half  = lane >> 4;
    const int batch = blockIdx.x >> 5;
    const int q0    = (blockIdx.x & 31) * 64 + wv * 16;
    const size_t brow = (size_t)batch * SEQ;

    unsigned short* Tk = &sK[wv * (32 * 64)];
    unsigned short* Tv = &sV[wv * (32 * 64)];
    unsigned short* P  = &sP[wv * (16 * 34)];

    const unsigned kb_lds = (unsigned)(size_t)(void*)Tk;
    const unsigned vb_lds = (unsigned)(size_t)(void*)Tv;

    const unsigned int* qw32 = (const unsigned int*)(qb + (brow + q0) * HDIM);

    // ---- load Q tile once: two A-fragments (d=0..31 and d=32..63)
    Frag qa[2];
#pragma unroll
    for (int f = 0; f < 2; ++f) {
#pragma unroll
        for (int j = 0; j < 8; ++j) {
            int t    = j * 32 + lane;       // 256 u32 = 16 rows x 16 words
            int rrow = t >> 4;
            int wd   = t & 15;
            unsigned int val = qw32[rrow * 32 + f * 16 + wd];
            *(unsigned int*)&P[rrow * 34 + 2 * wd] = val;
        }
        lds_fence();
#pragma unroll
        for (int j = 0; j < 8; ++j)
            qa[f].w[j] = *(const unsigned int*)&P[n * 34 + a_kidx(j, half)];
        lds_fence();   // WAR before restage / P reuse
    }

    v8f o_acc[4];
#pragma unroll
    for (int nt = 0; nt < 4; ++nt) o_acc[nt] = vzero8();
    float m_run[8], l_run[8];
#pragma unroll
    for (int r = 0; r < 8; ++r) { m_run[r] = -1e30f; l_run[r] = 0.f; }

    const int krow = lane >> 3;             // async tiling: 4 rows / 8 chunks
    const int kch  = lane & 7;

    const int nsteps = (q0 + 47) >> 5;      // causal: keys 0 .. q0+15
    for (int stp = 0; stp < nsteps; ++stp) {
        const int k0 = stp * 32;

        // All DS-gather data from the previous step has returned (WAR safety
        // vs the async engine, which is unordered with the DS queue).
        lds_fence();

        // ---- issue async K tile then V tile (8 x B128 each, 16B/lane)
#pragma unroll
        for (int it = 0; it < 8; ++it) {
            int row = it * 4 + krow;
            async_b128(kb_lds + row * 128 + kch * 16,
                       (unsigned long long)(size_t)
                           (kb + (brow + k0 + row) * HDIM + kch * 8));
        }
#pragma unroll
        for (int it = 0; it < 8; ++it) {
            int row = it * 4 + krow;
            async_b128(vb_lds + row * 128 + kch * 16,
                       (unsigned long long)(size_t)
                           (vb + (brow + k0 + row) * HDIM + kch * 8));
        }
        // Wait for the 8 K transfers only; V keeps flying during GEMM1.
        asm volatile("s_wait_asynccnt 0x8" ::: "memory");

        // ---- scores: 2 key-subtiles x 2 K-halves = 4 WMMAs
        v8f s[2];
#pragma unroll
        for (int sub = 0; sub < 2; ++sub) {
            v8f a = vzero8();
#pragma unroll
            for (int f = 0; f < 2; ++f) {
                Frag b;   // B[k=d][n=key] = K[key][d]
                const unsigned short* src =
                    &Tk[(sub * 16 + n) * 64 + f * 32 + half * 16];
#pragma unroll
                for (int j = 0; j < 8; ++j)
                    b.w[j] = *(const unsigned int*)&src[2 * j];
                a = __builtin_amdgcn_wmma_f32_16x16x32_bf16(
                    false, qa[f].v, false, b.v, (short)0, a, false, false);
            }
            s[sub] = a;
        }

        // ---- online softmax (row m = r + 8*half lives across 16 lanes)
        const bool edge = (k0 + 31 > q0);
#pragma unroll
        for (int r = 0; r < 8; ++r) {
            const int qrow = q0 + r + 8 * half;
            float e0 = s[0][r], e1 = s[1][r];
            if (edge) {
                if (k0 + n      > qrow) e0 = -1e30f;
                if (k0 + 16 + n > qrow) e1 = -1e30f;
            }
            float mx = fmaxf(e0, e1);
#pragma unroll
            for (int off = 8; off; off >>= 1)
                mx = fmaxf(mx, __shfl_xor(mx, off, 32));
            const float nm   = fmaxf(m_run[r], mx);
            const float corr = __expf(m_run[r] - nm);
            const float p0   = __expf(e0 - nm);
            const float p1   = __expf(e1 - nm);
            float rs = p0 + p1;
#pragma unroll
            for (int off = 8; off; off >>= 1)
                rs += __shfl_xor(rs, off, 32);
            l_run[r] = l_run[r] * corr + rs;
            m_run[r] = nm;
#pragma unroll
            for (int nt = 0; nt < 4; ++nt) o_acc[nt][r] *= corr;
            const int m = r + 8 * half;
            P[m * 34 + n]      = f2bf(p0);
            P[m * 34 + 16 + n] = f2bf(p1);
        }
        lds_fence();

        // ---- V resident, then O += P @ V (4 WMMAs over d-tiles)
        asm volatile("s_wait_asynccnt 0x0" ::: "memory");

        Frag pa;
#pragma unroll
        for (int j = 0; j < 8; ++j)
            pa.w[j] = *(const unsigned int*)&P[n * 34 + a_kidx(j, half)];
#pragma unroll
        for (int nt = 0; nt < 4; ++nt) {
            Frag b;   // B[k=key][n=d] = V[key][d], gathered down a d-column
            const int d = nt * 16 + n;
#pragma unroll
            for (int j = 0; j < 8; ++j) {
                int key = half * 16 + 2 * j;
                unsigned lo = Tv[key * 64 + d];
                unsigned hi = Tv[(key + 1) * 64 + d];
                b.w[j] = lo | (hi << 16);
            }
            o_acc[nt] = __builtin_amdgcn_wmma_f32_16x16x32_bf16(
                false, pa.v, false, b.v, (short)0, o_acc[nt], false, false);
        }

        if (stp + 1 < nsteps)   // pull next K tile toward L2/L0
            __builtin_prefetch(kb + (brow + k0 + 32) * HDIM, 0, 1);
    }

    // ---- epilogue: O / l, f32 out [row][64]
#pragma unroll
    for (int r = 0; r < 8; ++r) {
        const float inv = 1.0f / l_run[r];
        const int orow  = q0 + r + 8 * half;
        float* dst = out + (brow + orow) * HDIM;
#pragma unroll
        for (int nt = 0; nt < 4; ++nt)
            dst[nt * 16 + n] = o_acc[nt][r] * inv;
    }
}

// ---------------------------------------------------------------------------
extern "C" void kernel_launch(void* const* d_in, const int* in_sizes, int n_in,
                              void* d_out, int out_size, void* d_ws, size_t ws_size,
                              hipStream_t stream) {
    (void)in_sizes; (void)n_in; (void)out_size; (void)ws_size;
    const float* x  = (const float*)d_in[0];
    const float* Wq = (const float*)d_in[1];
    const float* Wk = (const float*)d_in[2];
    const float* Wv = (const float*)d_in[3];

    unsigned short* qb = (unsigned short*)d_ws;            // 16384x64 bf16
    unsigned short* kb = qb + (size_t)ROWS * HDIM;
    unsigned short* vb = kb + (size_t)ROWS * HDIM;
    float* o = (float*)d_out;

    qkv_proj_kernel<<<ROWS / 128, 256, 0, stream>>>(x, Wq, Wk, Wv, qb, kb, vb);
    attn_kernel<<<ROWS / 64, 128, 0, stream>>>(qb, kb, vb, o);
}
